// MoE_56470230008170
// MI455X (gfx1250) — matrix operand order repaired and checked by reading, not compiled, not run on hardware
//
#include <hip/hip_runtime.h>
#include <hip/hip_bf16.h>
#include <math.h>

// ---------------- Problem constants (from reference) ----------------
#define D_MODEL 1024      // D
#define E_EXP   8         // experts
#define HID     2048      // hidden (gate/value halves of 2*HID)
#define NTOK    8192      // 4*2048 tokens
#define NWAVES  8         // 256-thread block = 8 wave32
#define GSTRIDE (HID + 8)     // padded bf16 row stride for g-tile in LDS
#define XSTRIDE (D_MODEL + 8) // padded bf16 row stride for x-tile in LDS

typedef __attribute__((ext_vector_type(16))) __bf16 v16bf;
typedef __attribute__((ext_vector_type(8)))  __bf16 v8bf;
typedef __attribute__((ext_vector_type(8)))  float  v8f;

__device__ inline float fast_rcp(float v) { return __builtin_amdgcn_rcpf(v); }

// =====================================================================
// Kernel 1: router. One wave32 per token.
// =====================================================================
__global__ __launch_bounds__(256) void moe_router(
    const float* __restrict__ x, const float* __restrict__ Wr, const float* __restrict__ br,
    int* __restrict__ counts, float* __restrict__ psum, float* __restrict__ fcount,
    int* __restrict__ etok, float* __restrict__ escore)
{
  const int lane = threadIdx.x & 31;
  const int wave = threadIdx.x >> 5;
  const int n = blockIdx.x * NWAVES + wave;
  if (n >= NTOK) return;

  float acc[E_EXP];
#pragma unroll
  for (int e = 0; e < E_EXP; ++e) acc[e] = 0.f;

  const float* xr = x + (size_t)n * D_MODEL;
  for (int k = lane; k < D_MODEL; k += 32) {
    const float xv = xr[k];
    const float* wr = Wr + (size_t)k * E_EXP;
#pragma unroll
    for (int e = 0; e < E_EXP; ++e) acc[e] = fmaf(xv, wr[e], acc[e]);
  }
#pragma unroll
  for (int off = 16; off >= 1; off >>= 1) {
#pragma unroll
    for (int e = 0; e < E_EXP; ++e) acc[e] += __shfl_xor(acc[e], off, 32);
  }

  if (lane == 0) {
    float lg[E_EXP];
    float mx = -3.0e38f;
#pragma unroll
    for (int e = 0; e < E_EXP; ++e) { lg[e] = acc[e] + br[e]; mx = fmaxf(mx, lg[e]); }
    float pr[E_EXP]; float se = 0.f;
#pragma unroll
    for (int e = 0; e < E_EXP; ++e) { pr[e] = __expf(lg[e] - mx); se += pr[e]; }
    const float inv = fast_rcp(se);
#pragma unroll
    for (int e = 0; e < E_EXP; ++e) atomicAdd(&psum[e], pr[e] * inv);

    int i0 = 0;
#pragma unroll
    for (int e = 1; e < E_EXP; ++e) if (lg[e] > lg[i0]) i0 = e;
    int i1 = (i0 == 0) ? 1 : 0;
#pragma unroll
    for (int e = 0; e < E_EXP; ++e) if (e != i0 && lg[e] > lg[i1]) i1 = e;

    const float t  = __expf(lg[i1] - lg[i0]);   // <= 1
    const float s0 = fast_rcp(1.f + t);
    const float s1 = t * s0;

    atomicAdd(&fcount[i0], 1.f);
    atomicAdd(&fcount[i1], 1.f);

    int sl0 = atomicAdd(&counts[i0], 1);
    etok[i0 * NTOK + sl0] = n;  escore[i0 * NTOK + sl0] = s0;
    int sl1 = atomicAdd(&counts[i1], 1);
    etok[i1 * NTOK + sl1] = n;  escore[i1 * NTOK + sl1] = s1;
  }
}

// =====================================================================
// Kernel 2: aux loss finalize: E * sum_e (f[e]/N)*(p[e]/N)
// =====================================================================
__global__ void moe_aux(const float* __restrict__ psum, const float* __restrict__ fcount,
                        float* __restrict__ aux_out)
{
  if (threadIdx.x == 0 && blockIdx.x == 0) {
    float s = 0.f;
#pragma unroll
    for (int e = 0; e < E_EXP; ++e)
      s += (fcount[e] * (1.f / (float)NTOK)) * (psum[e] * (1.f / (float)NTOK));
    aux_out[0] = (float)E_EXP * s;
  }
}

// =====================================================================
// Kernel 3: weight repack f32 -> bf16, fragment-contiguous.
//   W viewed as [ktot][ldb]; packed fragment = 32 consecutive k per column,
//   so a lane's WMMA B fragment is ONE contiguous 32-byte load.
// =====================================================================
__global__ __launch_bounds__(256) void pack_weights(
    const float* __restrict__ W, __bf16* __restrict__ Wp, int ldb, int ktot)
{
  const size_t idx = (size_t)blockIdx.x * 256 + threadIdx.x;   // fragment id
  const size_t nfrag = (size_t)(ktot / 32) * (size_t)ldb;
  if (idx >= nfrag) return;
  const int    col = (int)(idx % (size_t)ldb);
  const size_t gk  = (idx / (size_t)ldb) * 32;

  v16bf v0, v1;
#pragma unroll
  for (int i = 0; i < 16; ++i) v0[i] = (__bf16)W[(gk + i) * (size_t)ldb + col];
#pragma unroll
  for (int i = 0; i < 16; ++i) v1[i] = (__bf16)W[(gk + 16 + i) * (size_t)ldb + col];
  v16bf* dst = (v16bf*)(Wp + idx * 32);   // 64B aligned, coalesced across threads
  dst[0] = v0;
  dst[1] = v1;
}

// ---- WMMA fragment helpers (CDNA5 wave32, 16x16x32 bf16; contiguous k in A/B, same order in both) ----
// A (16x32): lane row m = lane&15; a[i] -> K = 16*half+i
// B (32x16): b[i] = B[16*half + i][lane&15]
// C (16x16): c[r] = C[r + 8*half][lane&15]
__device__ inline v16bf combine16(v8bf lo, v8bf hi) {
  v16bf a;
#pragma unroll
  for (int i = 0; i < 8; ++i) { a[i] = lo[i]; a[8 + i] = hi[i]; }
  return a;
}

__device__ inline v16bf load_a_lds(const __bf16* base, int stride, int row, int k0, int half) {
  const v8bf lo = *(const v8bf*)(base + (size_t)row * stride + k0 + 16 * half);
  const v8bf hi = *(const v8bf*)(base + (size_t)row * stride + k0 + 16 * half + 8);
  return combine16(lo, hi);
}

__device__ inline v16bf load_b_packed(const __bf16* __restrict__ Wp, int ldb,
                                      int gkblk, int col, int half) {
  return *(const v16bf*)(Wp + ((size_t)gkblk * ldb + col) * 32 + half * 16);
}

__device__ inline v16bf load_b_direct(const float* __restrict__ W, int ldb,
                                      int k0, int col, int half) {
  const float* wp = W + (size_t)(k0 + 16 * half) * ldb + col;
  v16bf b;
#pragma unroll
  for (int i = 0; i < 16; ++i) b[i] = (__bf16)wp[(size_t)i * ldb];
  return b;
}

#define WMMA_BF16(A, B, C) \
  __builtin_amdgcn_wmma_f32_16x16x32_bf16(false, (A), false, (B), (short)0, (C), false, false)

// =====================================================================
// Kernel 4: fused expert FFN for a 16-token tile of one expert.
//   Each wave owns 32 output columns per iteration: one A fragment feeds
//   4 WMMAs in GEMM1 (2 gate cols x paired value cols) and 2 in GEMM2.
// =====================================================================
template <bool PACKED>
__global__ __launch_bounds__(256) void moe_expert_ffn(
    const float* __restrict__ x,
    const float* __restrict__ W1, const __bf16* __restrict__ W1p, const float* __restrict__ b1,
    const float* __restrict__ W2, const __bf16* __restrict__ W2p, const float* __restrict__ b2,
    const int* __restrict__ counts, const int* __restrict__ etok,
    const float* __restrict__ escore, float* __restrict__ out)
{
  extern __shared__ char smem_raw[];
  __bf16* g_lds   = (__bf16*)smem_raw;                  // 16 x GSTRIDE
  __bf16* x_lds   = g_lds + 16 * GSTRIDE;               // 16 x XSTRIDE
  int*    s_rows  = (int*)(x_lds + 16 * XSTRIDE);
  float*  s_score = (float*)(s_rows + 16);

  const int e     = blockIdx.y;
  const int cnt   = counts[e];
  const int tile0 = blockIdx.x * 16;
  if (tile0 >= cnt) return;   // over-provisioned grid: dead tiles exit

  const int tid  = threadIdx.x;
  const int lane = tid & 31;
  const int wave = tid >> 5;
  const int half = lane >> 4;
  const int col  = lane & 15;

  if (tid < 16) {
    const int idx = tile0 + tid;
    if (idx < cnt) { s_rows[tid] = etok[e * NTOK + idx]; s_score[tid] = escore[e * NTOK + idx]; }
    else           { s_rows[tid] = 0;                    s_score[tid] = 0.f; }
  }
  __syncthreads();

  // ---- stage x tile (16 x D_MODEL) to LDS as bf16, converted once ----
  for (int it = tid; it < 16 * (D_MODEL / 16); it += 256) {
    const int row = it >> 6;                  // D_MODEL/16 = 64 chunks per row
    const int cc  = (it & 63) * 16;
    const float* p = x + (size_t)s_rows[row] * D_MODEL + cc;
    float4 f0 = *(const float4*)(p);
    float4 f1 = *(const float4*)(p + 4);
    float4 f2 = *(const float4*)(p + 8);
    float4 f3 = *(const float4*)(p + 12);
    v16bf v;
    v[0]=(__bf16)f0.x;  v[1]=(__bf16)f0.y;  v[2]=(__bf16)f0.z;  v[3]=(__bf16)f0.w;
    v[4]=(__bf16)f1.x;  v[5]=(__bf16)f1.y;  v[6]=(__bf16)f1.z;  v[7]=(__bf16)f1.w;
    v[8]=(__bf16)f2.x;  v[9]=(__bf16)f2.y;  v[10]=(__bf16)f2.z; v[11]=(__bf16)f2.w;
    v[12]=(__bf16)f3.x; v[13]=(__bf16)f3.y; v[14]=(__bf16)f3.z; v[15]=(__bf16)f3.w;
    *(v16bf*)&x_lds[(size_t)row * XSTRIDE + cc] = v;
  }
  __syncthreads();

  const float* W1e  = W1 + (size_t)e * D_MODEL * (2 * HID);
  const float* b1e  = b1 + (size_t)e * (2 * HID);

  // -------- GEMM1: 32 gate cols + paired value cols per wave-iter ------
  for (int t = wave; t < HID / 32; t += NWAVES) {
    const int hc = t * 32;
    v8f acc_a0 = {}, acc_a1 = {};  // gate halves  (cols hc, hc+16)
    v8f acc_b0 = {}, acc_b1 = {};  // value halves (cols HID+hc, HID+hc+16)
    for (int k0 = 0; k0 < D_MODEL; k0 += 32) {
      const v16bf a = load_a_lds(x_lds, XSTRIDE, col, k0, half);
      v16bf ba0, ba1, bb0, bb1;
      if (PACKED) {
        const int gkblk = e * (D_MODEL / 32) + (k0 >> 5);
        ba0 = load_b_packed(W1p, 2 * HID, gkblk, hc + col, half);
        ba1 = load_b_packed(W1p, 2 * HID, gkblk, hc + 16 + col, half);
        bb0 = load_b_packed(W1p, 2 * HID, gkblk, HID + hc + col, half);
        bb1 = load_b_packed(W1p, 2 * HID, gkblk, HID + hc + 16 + col, half);
        if (k0 + 32 < D_MODEL)
          __builtin_prefetch(W1p + ((size_t)(gkblk + 1) * (2 * HID) + hc + col) * 32, 0, 1);
      } else {
        ba0 = load_b_direct(W1e, 2 * HID, k0, hc + col, half);
        ba1 = load_b_direct(W1e, 2 * HID, k0, hc + 16 + col, half);
        bb0 = load_b_direct(W1e, 2 * HID, k0, HID + hc + col, half);
        bb1 = load_b_direct(W1e, 2 * HID, k0, HID + hc + 16 + col, half);
      }
      acc_a0 = WMMA_BF16(a, ba0, acc_a0);
      acc_a1 = WMMA_BF16(a, ba1, acc_a1);
      acc_b0 = WMMA_BF16(a, bb0, acc_b0);
      acc_b1 = WMMA_BF16(a, bb1, acc_b1);
    }
    const float bva0 = b1e[hc + col],      bvb0 = b1e[HID + hc + col];
    const float bva1 = b1e[hc + 16 + col], bvb1 = b1e[HID + hc + 16 + col];
#pragma unroll
    for (int r = 0; r < 8; ++r) {
      const int row = r + 8 * half;
      const float a0 = acc_a0[r] + bva0, v0 = acc_b0[r] + bvb0;
      const float a1 = acc_a1[r] + bva1, v1 = acc_b1[r] + bvb1;
      g_lds[(size_t)row * GSTRIDE + hc + col]      = (__bf16)(a0 * fast_rcp(1.f + __expf(-a0)) * v0);
      g_lds[(size_t)row * GSTRIDE + hc + 16 + col] = (__bf16)(a1 * fast_rcp(1.f + __expf(-a1)) * v1);
    }
  }
  __syncthreads();

  // -------- GEMM2: y = g @ W2[e] + b2 ; scaled atomic scatter ----------
  const float* W2e = W2 + (size_t)e * HID * D_MODEL;
  const float* b2e = b2 + (size_t)e * D_MODEL;

  for (int t = wave; t < D_MODEL / 32; t += NWAVES) {
    const int n0 = t * 32;
    v8f acc0 = {}, acc1 = {};
    for (int k0 = 0; k0 < HID; k0 += 32) {
      const v16bf a = load_a_lds(g_lds, GSTRIDE, col, k0, half);
      v16bf b0, b1f;
      if (PACKED) {
        const int gkblk = e * (HID / 32) + (k0 >> 5);
        b0  = load_b_packed(W2p, D_MODEL, gkblk, n0 + col, half);
        b1f = load_b_packed(W2p, D_MODEL, gkblk, n0 + 16 + col, half);
        if (k0 + 32 < HID)
          __builtin_prefetch(W2p + ((size_t)(gkblk + 1) * D_MODEL + n0 + col) * 32, 0, 1);
      } else {
        b0  = load_b_direct(W2e, D_MODEL, k0, n0 + col, half);
        b1f = load_b_direct(W2e, D_MODEL, k0, n0 + 16 + col, half);
      }
      acc0 = WMMA_BF16(a, b0, acc0);
      acc1 = WMMA_BF16(a, b1f, acc1);
    }
    const float b2v0 = b2e[n0 + col];
    const float b2v1 = b2e[n0 + 16 + col];
#pragma unroll
    for (int r = 0; r < 8; ++r) {
      const int row = r + 8 * half;
      if (tile0 + row < cnt) {
        float* op = out + (size_t)s_rows[row] * D_MODEL + n0;
        const float sc = s_score[row];
        atomicAdd(op + col,      sc * (acc0[r] + b2v0));
        atomicAdd(op + 16 + col, sc * (acc1[r] + b2v1));
      }
    }
  }
}

// =====================================================================
// Host launcher
// =====================================================================
extern "C" void kernel_launch(void* const* d_in, const int* in_sizes, int n_in,
                              void* d_out, int out_size, void* d_ws, size_t ws_size,
                              hipStream_t stream) {
  (void)in_sizes; (void)n_in;
  const float* x  = (const float*)d_in[0];
  const float* Wr = (const float*)d_in[1];
  const float* br = (const float*)d_in[2];
  const float* W1 = (const float*)d_in[3];
  const float* b1 = (const float*)d_in[4];
  const float* W2 = (const float*)d_in[5];
  const float* b2 = (const float*)d_in[6];
  float* out = (float*)d_out;

  // ws layout (4B units): [0..7] counts(int) [8..15] psum [16..23] fcount
  //   [32..) etok int[E*NTOK] ; then escore f32[E*NTOK] ; then packed weights.
  int*   counts = (int*)d_ws;
  float* psum   = (float*)d_ws + 8;
  float* fcount = (float*)d_ws + 16;
  int*   etok   = (int*)d_ws + 32;
  float* escore = (float*)d_ws + 32 + E_EXP * NTOK;

  const size_t listBytes = (size_t)(32 + 2 * E_EXP * NTOK) * 4;
  const size_t packOff   = (listBytes + 255) & ~(size_t)255;
  const size_t w1pElems  = (size_t)E_EXP * D_MODEL * (2 * HID);   // 33.5M bf16
  const size_t w2pElems  = (size_t)E_EXP * HID * D_MODEL;         // 16.8M bf16
  __bf16* W1p = (__bf16*)((char*)d_ws + packOff);
  __bf16* W2p = W1p + w1pElems;
  const bool packed = ws_size >= packOff + (w1pElems + w2pElems) * sizeof(__bf16);

  hipMemsetAsync(d_ws, 0, 128, stream);                               // counts/psum/fcount
  hipMemsetAsync(d_out, 0, (size_t)out_size * sizeof(float), stream); // out accumulated atomically

  moe_router<<<NTOK / NWAVES, 256, 0, stream>>>(x, Wr, br, counts, psum, fcount, etok, escore);
  moe_aux<<<1, 32, 0, stream>>>(psum, fcount, out + (size_t)NTOK * D_MODEL);

  const dim3 grid(NTOK / 16, E_EXP);  // over-provisioned; blocks past counts[e] exit
  const size_t smem = (size_t)16 * GSTRIDE * sizeof(__bf16)
                    + (size_t)16 * XSTRIDE * sizeof(__bf16)
                    + 16 * sizeof(int) + 16 * sizeof(float);

  if (packed) {
    const size_t nfrag1 = (size_t)(E_EXP * D_MODEL / 32) * (2 * HID);
    const size_t nfrag2 = (size_t)(E_EXP * HID / 32) * D_MODEL;
    pack_weights<<<(unsigned)((nfrag1 + 255) / 256), 256, 0, stream>>>(W1, W1p, 2 * HID, E_EXP * D_MODEL);
    pack_weights<<<(unsigned)((nfrag2 + 255) / 256), 256, 0, stream>>>(W2, W2p, D_MODEL, E_EXP * HID);
    moe_expert_ffn<true><<<grid, 256, smem, stream>>>(x, W1, W1p, b1, W2, W2p, b2,
                                                      counts, etok, escore, out);
  } else {
    moe_expert_ffn<false><<<grid, 256, smem, stream>>>(x, W1, W1p, b1, W2, W2p, b2,
                                                       counts, etok, escore, out);
  }
}
